// RobustSymGAT_26053271617751
// MI455X (gfx1250) — matrix operand order repaired and checked by reading, not compiled, not run on hardware
//
#include <hip/hip_runtime.h>
#include <math.h>

#define N_NODES 8192
#define N_EDGES 262144
#define IN_DIM  16
#define HDIM    64
#define HALFD   32
#define SYMD    10
#define EPSF    1e-8f

typedef __attribute__((ext_vector_type(16))) _Float16 v16h;
typedef __attribute__((ext_vector_type(8)))  _Float16 v8h;
typedef __attribute__((ext_vector_type(8)))  float    v8f;

__device__ __forceinline__ float eluf(float x) { return x > 0.f ? x : (__expf(x) - 1.f); }

// penalty with finite hi
__device__ __forceinline__ float pen2(float v, float lo, float hi) {
    float low  = (lo - v) / lo;
    float high = (v - hi) / hi;
    return v < lo ? low : (v > hi ? high : 0.f);
}

// monotonic float<->uint map for atomicMax on floats of any sign
__device__ __forceinline__ unsigned mapF(float v) {
    unsigned u = __float_as_uint(v);
    return (u & 0x80000000u) ? ~u : (u | 0x80000000u);
}
__device__ __forceinline__ float unmapF(unsigned u) {
    return (u & 0x80000000u) ? __uint_as_float(u & 0x7fffffffu) : __uint_as_float(~u);
}

// ---------------------------------------------------------------------------
// Weight prep: transpose GAT / head weights into k-major f16 (pad layer0 K 48->64)
// ---------------------------------------------------------------------------
__global__ void prep_weights(const float* __restrict__ gW0, const float* __restrict__ gW1,
                             const float* __restrict__ gW2, const float* __restrict__ rW1,
                             const float* __restrict__ cW1,
                             _Float16* __restrict__ Wt0, _Float16* __restrict__ Wt1,
                             _Float16* __restrict__ Wt2, _Float16* __restrict__ rWt,
                             _Float16* __restrict__ cWt) {
    int idx = blockIdx.x * blockDim.x + threadIdx.x;
    if (idx < 4096) {                       // Wt0: 64x64, K padded (gat_W0 is 64x48)
        int k = idx >> 6, n = idx & 63;
        Wt0[idx] = (k < 48) ? (_Float16)gW0[n * 48 + k] : (_Float16)0.f;
    } else if (idx < 8192) {                // Wt1
        int i = idx - 4096; int k = i >> 6, n = i & 63;
        Wt1[i] = (_Float16)gW1[n * 64 + k];
    } else if (idx < 12288) {               // Wt2
        int i = idx - 8192; int k = i >> 6, n = i & 63;
        Wt2[i] = (_Float16)gW2[n * 64 + k];
    } else if (idx < 14336) {               // reg_W1: 32x64 -> 64x32 k-major
        int i = idx - 12288; int k = i >> 5, n = i & 31;
        rWt[i] = (_Float16)rW1[n * 64 + k];
    } else if (idx < 16384) {               // cls_W1
        int i = idx - 14336; int k = i >> 5, n = i & 31;
        cWt[i] = (_Float16)cW1[n * 64 + k];
    }
}

// ---------------------------------------------------------------------------
// Symbolic features + sym encoder + concat -> h (N x 64, cols 48..63 zero pad)
// ---------------------------------------------------------------------------
__global__ void features_kernel(const float* __restrict__ x, const float* __restrict__ symW,
                                const float* __restrict__ symb,
                                float* __restrict__ h, _Float16* __restrict__ h16) {
    int i = blockIdx.x * blockDim.x + threadIdx.x;
    if (i >= N_NODES) return;
    const float* xi = x + i * IN_DIM;
    float am = xi[0], bod = xi[1], dox = xi[2], ph = xi[4], ni = xi[7];
    float p_ph  = pen2(ph, 6.5f, 8.5f);
    float p_am  = pen2(am, 0.001f, 0.5f);
    float p_bod = pen2(bod, 0.001f, 5.0f);
    float p_do  = (dox < 6.0f) ? (6.0f - dox) / 6.0f : 0.f;   // hi = inf
    float p_ni  = pen2(ni, 0.001f, 10.0f);
    float bact = (1.2f * am / (0.5f + EPSF) + 1.5f * bod / (5.0f + EPSF)
                  - 0.8f * dox / (10.0f + EPSF)) / 3.0f;
    float chem = (1.5f * ph / (8.5f + EPSF) + 1.0f * ni / (10.0f + EPSF)) / 2.0f;
    float org  = (2.0f * bod / (5.0f + EPSF) - 1.5f * dox / (10.0f + EPSF)
                  + 0.8f * am / (0.5f + EPSF)) / 3.0f;
    float agri = 2.0f * ni / (10.0f + EPSF);
    float comp = 1.0f / (1.0f + (p_ph + p_am + p_bod + p_do + p_ni) + EPSF);
    float sym[SYMD] = {p_ph, p_am, p_bod, p_do, p_ni, bact, chem, org, agri, comp};

    float* hi = h + i * HDIM;
    _Float16* h16i = h16 + i * HDIM;
#pragma unroll
    for (int k = 0; k < IN_DIM; ++k) { hi[k] = xi[k]; h16i[k] = (_Float16)xi[k]; }
    for (int j = 0; j < HALFD; ++j) {
        float acc = symb[j];
#pragma unroll
        for (int s = 0; s < SYMD; ++s) acc += sym[s] * symW[j * SYMD + s];
        acc = eluf(acc);
        hi[IN_DIM + j] = acc; h16i[IN_DIM + j] = (_Float16)acc;
    }
#pragma unroll
    for (int k = IN_DIM + HALFD; k < HDIM; ++k) { hi[k] = 0.f; h16i[k] = (_Float16)0.f; }
}

// ---------------------------------------------------------------------------
// WMMA GEMM: C[M x NCOLS] = A[M x 64] (f16) * Bt[64 x NCOLS] (f16, k-major)
// block = 128 threads (4 waves), computes 64 rows x 16 cols; B staged in LDS.
// NCOLS is compile-time so the 8 D-layout stores fold into immediate offsets.
// ---------------------------------------------------------------------------
template <bool BIAS_ELU, int NCOLS>
__global__ void gemm16(const _Float16* __restrict__ A, const _Float16* __restrict__ Bt,
                       const float* __restrict__ bias, float* __restrict__ C) {
    __shared__ __align__(16) _Float16 sB[64 * 16];
    const int n0 = blockIdx.y * 16;
    const int row0 = blockIdx.x * 64;
    const int t = threadIdx.x;
    for (int idx = t; idx < 64 * 16; idx += 128) {
        int k = idx >> 4, n = idx & 15;
        sB[idx] = Bt[k * NCOLS + n0 + n];
    }
    __syncthreads();

    const int lane = t & 31;
    const int wave = t >> 5;
    const int half = lane >> 4;   // lane group
    const int l15  = lane & 15;
    const int rowA = row0 + wave * 16 + l15;

    v8f acc = {};
#pragma unroll
    for (int ks = 0; ks < 64; ks += 32) {
        // A 16x32 f16 layout: lane (l15, half): K = ks + half*8 + [0..7], and +16
        const _Float16* ap = A + rowA * 64 + ks + half * 8;
        v8h alo = *(const v8h*)ap;
        v8h ahi = *(const v8h*)(ap + 16);
        v16h a;
#pragma unroll
        for (int q = 0; q < 8; ++q) { a[q] = alo[q]; a[q + 8] = ahi[q]; }
        // B 32x16 f16 layout: lane (l15, half): col N = l15, K = ks + half*8 + [0..7], and +16
        const _Float16* bp = &sB[(ks + half * 8) * 16 + l15];
        v16h b;
#pragma unroll
        for (int q = 0; q < 8; ++q) { b[q] = bp[q * 16]; b[q + 8] = bp[(q + 16) * 16]; }
        acc = __builtin_amdgcn_wmma_f32_16x16x32_f16(false, a, false, b, (short)0, acc,
                                                     false, false);
    }
    // D layout: lane L -> col n0 + (L&15); VGPR v -> row wave*16 + (L>>4)*8 + v
    const int ccol = n0 + l15;
    float bval = BIAS_ELU ? bias[ccol] : 0.f;
    float* cp = C + (size_t)(row0 + wave * 16 + half * 8) * NCOLS + ccol;
#pragma unroll
    for (int v = 0; v < 8; ++v) {
        float val = acc[v];
        if (BIAS_ELU) val = eluf(val + bval);
        cp[v * NCOLS] = val;
    }
}

// ---------------------------------------------------------------------------
// Attention scalars per node: a_src = ht.att[:64], a_dst = ht.att[64:]
// ---------------------------------------------------------------------------
__global__ void attn_node(const float* __restrict__ ht, const float* __restrict__ att,
                          float* __restrict__ asrc, float* __restrict__ adst) {
    int i = blockIdx.x * blockDim.x + threadIdx.x;
    if (i >= N_NODES) return;
    const float* hi = ht + i * HDIM;
    float s = 0.f, d = 0.f;
#pragma unroll
    for (int k = 0; k < HDIM; ++k) { float v = hi[k]; s += v * att[k]; d += v * att[HDIM + k]; }
    asrc[i] = s; adst[i] = d;
}

// per edge: leaky_relu score + global max (monotonic-uint atomicMax)
__global__ void edge_kernel(const int* __restrict__ ei, const float* __restrict__ asrc,
                            const float* __restrict__ adst, float* __restrict__ ebuf,
                            unsigned* __restrict__ gmax) {
    int k = blockIdx.x * blockDim.x + threadIdx.x;
    if (k >= N_EDGES) return;
    __builtin_prefetch(&ei[k + 2048], 0, 1);             // global_prefetch, next tiles
    __builtin_prefetch(&ei[N_EDGES + k + 2048], 0, 1);
    int r = ei[k], c = ei[N_EDGES + k];
    float e = asrc[r] + adst[c];
    e = e > 0.f ? e : 0.2f * e;
    ebuf[k] = e;
    atomicMax(gmax, mapF(e));
}

// colsum of ht (background term of the softmax-decomposed matmul)
__global__ void colsum_kernel(const float* __restrict__ ht, float* __restrict__ csum) {
    int j = threadIdx.x;                    // 64 threads = 64 columns
    int r0 = blockIdx.x * 128;
    float s = 0.f;
    for (int rr = 0; rr < 128; ++rr) s += ht[(r0 + rr) * HDIM + j];
    atomicAdd(&csum[j], s);
}

// sparse correction scatter: symmetrized edge contributes w = e^{v-m} - e^{-m}
// at (r,c) and (c,r). 64 threads per edge (one per feature column).
__global__ void scatter_kernel(const int* __restrict__ ei, const float* __restrict__ ebuf,
                               const float* __restrict__ ht, const unsigned* __restrict__ gmax,
                               float* __restrict__ numer, float* __restrict__ dcorr) {
    int tid = blockIdx.x * blockDim.x + threadIdx.x;
    int k = tid >> 6;
    int j = tid & 63;
    if (k >= N_EDGES) return;
    if (j == 32) {                                       // one lane prefetches ahead
        __builtin_prefetch(&ei[k + 64], 0, 1);
        __builtin_prefetch(&ei[N_EDGES + k + 64], 0, 1);
        __builtin_prefetch(&ebuf[k + 64], 0, 1);
    }
    int r = ei[k], c = ei[N_EDGES + k];
    float m  = unmapF(*gmax);
    float em = __expf(-m);
    float v  = 0.5f * ebuf[k];              // (A + A^T)/2 contribution
    float w  = __expf(v - m) - em;
    atomicAdd(&numer[r * HDIM + j], w * ht[c * HDIM + j]);
    atomicAdd(&numer[c * HDIM + j], w * ht[r * HDIM + j]);
    if (j == 0) { atomicAdd(&dcorr[r], w); atomicAdd(&dcorr[c], w); }
}

// h_new = (residual? h_old : 0) + elu( (em*colsum + numer + (1-em)*ht) / denom )
__global__ void finalize_kernel(const float* __restrict__ ht, const float* __restrict__ numer,
                                const float* __restrict__ dcorr, const float* __restrict__ csum,
                                const unsigned* __restrict__ gmax, int residual,
                                float* __restrict__ h, _Float16* __restrict__ h16) {
    int tid = blockIdx.x * blockDim.x + threadIdx.x;
    if (tid >= N_NODES * HDIM) return;
    int i = tid >> 6, j = tid & 63;
    float m  = unmapF(*gmax);
    float em = __expf(-m);
    float dw = 1.0f - em;                   // diagonal (value m) correction
    float denom = (float)N_NODES * em + dcorr[i] + dw;
    float num = em * csum[j] + numer[tid] + dw * ht[tid];
    float a = eluf(num / denom);
    float outv = residual ? (h[tid] + a) : a;
    h[tid] = outv;
    h16[tid] = (_Float16)outv;
}

// head projections + output packing: [preds N][logits N*4][emb N*64]
__global__ void heads_final(const float* __restrict__ treg, const float* __restrict__ tcls,
                            const float* __restrict__ rW2, const float* __restrict__ rb2,
                            const float* __restrict__ cW2, const float* __restrict__ cb2,
                            const float* __restrict__ h, float* __restrict__ out) {
    int i = blockIdx.x * blockDim.x + threadIdx.x;
    if (i >= N_NODES) return;
    float p = rb2[0];
#pragma unroll
    for (int k = 0; k < HALFD; ++k) p += treg[i * HALFD + k] * rW2[k];
    out[i] = p;
#pragma unroll
    for (int c = 0; c < 4; ++c) {
        float l = cb2[c];
#pragma unroll
        for (int k = 0; k < HALFD; ++k) l += tcls[i * HALFD + k] * cW2[c * HALFD + k];
        out[N_NODES + i * 4 + c] = l;
    }
#pragma unroll
    for (int j = 0; j < HDIM; ++j) out[5 * N_NODES + i * HDIM + j] = h[i * HDIM + j];
}

// ---------------------------------------------------------------------------
extern "C" void kernel_launch(void* const* d_in, const int* in_sizes, int n_in,
                              void* d_out, int out_size, void* d_ws, size_t ws_size,
                              hipStream_t stream) {
    (void)in_sizes; (void)n_in; (void)out_size; (void)ws_size;
    const float* x    = (const float*)d_in[0];
    const int*   ei   = (const int*)d_in[1];
    /* d_in[2] seed_indices unused by reference */
    const float* symW = (const float*)d_in[3];
    const float* symb = (const float*)d_in[4];
    const float* gW0  = (const float*)d_in[5];
    const float* att0 = (const float*)d_in[6];
    const float* gW1  = (const float*)d_in[7];
    const float* att1 = (const float*)d_in[8];
    const float* gW2  = (const float*)d_in[9];
    const float* att2 = (const float*)d_in[10];
    const float* rW1  = (const float*)d_in[11];
    const float* rb1  = (const float*)d_in[12];
    const float* rW2  = (const float*)d_in[13];
    const float* rb2  = (const float*)d_in[14];
    const float* cW1  = (const float*)d_in[15];
    const float* cb1  = (const float*)d_in[16];
    const float* cW2  = (const float*)d_in[17];
    const float* cb2  = (const float*)d_in[18];
    float* out = (float*)d_out;

    char* ws = (char*)d_ws;
    size_t off = 0;
    auto carve = [&](size_t bytes) -> char* {
        off = (off + 255) & ~(size_t)255;
        char* p = ws + off; off += bytes; return p;
    };
    float*    h    = (float*)carve((size_t)N_NODES * HDIM * 4);
    _Float16* h16  = (_Float16*)carve((size_t)N_NODES * HDIM * 2);
    float*    ht   = (float*)carve((size_t)N_NODES * HDIM * 4);
    _Float16* Wt0  = (_Float16*)carve(64 * 64 * 2);
    _Float16* Wt1  = (_Float16*)carve(64 * 64 * 2);
    _Float16* Wt2  = (_Float16*)carve(64 * 64 * 2);
    _Float16* rWt  = (_Float16*)carve(64 * 32 * 2);
    _Float16* cWt  = (_Float16*)carve(64 * 32 * 2);
    float*    asrc = (float*)carve((size_t)N_NODES * 4);
    float*    adst = (float*)carve((size_t)N_NODES * 4);
    float*    ebuf = (float*)carve((size_t)N_EDGES * 4);
    unsigned* gmax = (unsigned*)carve(4);
    float*    csum = (float*)carve(HDIM * 4);
    float*    dcorr= (float*)carve((size_t)N_NODES * 4);
    float*    numer= (float*)carve((size_t)N_NODES * HDIM * 4);
    float*    treg = (float*)carve((size_t)N_NODES * HALFD * 4);
    float*    tcls = (float*)carve((size_t)N_NODES * HALFD * 4);

    prep_weights<<<64, 256, 0, stream>>>(gW0, gW1, gW2, rW1, cW1, Wt0, Wt1, Wt2, rWt, cWt);
    features_kernel<<<N_NODES / 256, 256, 0, stream>>>(x, symW, symb, h, h16);

    const _Float16* Wts[3]  = {Wt0, Wt1, Wt2};
    const float*    atts[3] = {att0, att1, att2};
    for (int layer = 0; layer < 3; ++layer) {
        gemm16<false, HDIM><<<dim3(N_NODES / 64, HDIM / 16), 128, 0, stream>>>(
            h16, Wts[layer], nullptr, ht);
        attn_node<<<N_NODES / 256, 256, 0, stream>>>(ht, atts[layer], asrc, adst);
        hipMemsetAsync(gmax, 0, 4, stream);
        hipMemsetAsync(csum, 0, HDIM * 4, stream);
        hipMemsetAsync(dcorr, 0, (size_t)N_NODES * 4, stream);
        hipMemsetAsync(numer, 0, (size_t)N_NODES * HDIM * 4, stream);
        edge_kernel<<<N_EDGES / 256, 256, 0, stream>>>(ei, asrc, adst, ebuf, gmax);
        colsum_kernel<<<N_NODES / 128, 64, 0, stream>>>(ht, csum);
        scatter_kernel<<<(N_EDGES * 64) / 256, 256, 0, stream>>>(ei, ebuf, ht, gmax, numer, dcorr);
        finalize_kernel<<<(N_NODES * HDIM) / 256, 256, 0, stream>>>(
            ht, numer, dcorr, csum, gmax, layer > 0 ? 1 : 0, h, h16);
    }
    gemm16<true, HALFD><<<dim3(N_NODES / 64, HALFD / 16), 128, 0, stream>>>(h16, rWt, rb1, treg);
    gemm16<true, HALFD><<<dim3(N_NODES / 64, HALFD / 16), 128, 0, stream>>>(h16, cWt, cb1, tcls);
    heads_final<<<N_NODES / 256, 256, 0, stream>>>(treg, tcls, rW2, rb2, cW2, cb2, h, out);
}
